// HAMLET_14001593385707
// MI455X (gfx1250) — matrix operand order repaired and checked by reading, not compiled, not run on hardware
//
#include <hip/hip_runtime.h>
#include <math.h>

typedef _Float16 v16h __attribute__((ext_vector_type(16)));
typedef float    v8f  __attribute__((ext_vector_type(8)));
typedef _Float16 h4   __attribute__((ext_vector_type(4)));
typedef _Float16 h8   __attribute__((ext_vector_type(8)));

#define BM  128
#define BN  64
#define BKK 32

#define GF_AT   1   // A stored [K,M]: A[k*lda + m]
#define GF_BROW 2   // B stored row-major [K,N]: B[k*ldb + n]; default N-major B[n*ldb + k]
#define GF_ACC  4   // accumulate into existing C before activation
#define GF_RELU 8
#define GF_SIG  16

// ---------------------------------------------------------------------------
// Generic WMMA GEMM: C = act(alpha * (A*B) + bias + (ACC ? C : 0))
// 256 threads = 8 waves (4x2); each wave owns a 32x32 output tile -> four
// v_wmma_f32_16x16x32_f16 per 32-deep k-step. Double-buffered LDS with
// explicit load/compute/store pipelining: tile k+1's global loads issue
// before tile k's WMMAs, LDS stores drain afterwards. VA/VB (float4-along-K
// staging) are template params so each instantiation has one clean path.
// M/N handled by index clamping; K zero-padded via a statically-separate
// tail step (clamped-address loads + zero-select, never predicated loads).
// ---------------------------------------------------------------------------
template <int FLAGS, bool VA, bool VB>
__global__ __launch_bounds__(256)
void hamlet_gemm_wmma(const float* __restrict__ A, const float* __restrict__ Bm,
                      const float* __restrict__ bias, float* __restrict__ C,
                      int M, int N, int K, int lda, int ldb, int ldc,
                      long sAz, long sBz, long sCz, float alpha)
{
    __shared__ __attribute__((aligned(16))) _Float16 As[2][BM * BKK];  // [m][k]
    __shared__ __attribute__((aligned(16))) _Float16 Bs[2][BN * BKK];  // [n][k]
    const int z = blockIdx.z;
    A  += (long)z * sAz;
    Bm += (long)z * sBz;
    C  += (long)z * sCz;
    const int tile_m = blockIdx.y * BM;
    const int tile_n = blockIdx.x * BN;
    const int tid  = threadIdx.x;
    const int lane = tid & 31;
    const int wid  = tid >> 5;
    const int wm   = wid >> 1;   // 0..3  (32-row strip)
    const int wn   = wid & 1;    // 0..1  (32-col strip)
    const int Mc = M - 1, Nc = N - 1;

    v8f acc00 = {}, acc01 = {}, acc10 = {}, acc11 = {};

    // staging registers (dead arrays eliminated per instantiation)
    float4 rva4[4];
    float  rva[16];
    float4 rvb4[2];
    float  rvb[8];

    // Phase 1: issue all global loads for one 32-deep k-slice into registers.
    auto stage_load = [&](int k0, bool tailk) {
        if (FLAGS & GF_AT) {
            #pragma unroll
            for (int p = 0; p < 16; ++p) {
                int e = (p << 8) + tid;
                int m = e >> 5, k = e & 31;
                int gm = tile_m + m; gm = gm > Mc ? Mc : gm;
                int gk = k0 + k;
                if (tailk && gk > K - 1) gk = K - 1;   // clamp: load stays unconditional
                rva[p] = A[(long)gk * lda + gm];
            }
        } else if (VA) {
            #pragma unroll
            for (int p = 0; p < 4; ++p) {
                int e = (p << 8) + tid;
                int m = e >> 3, kq = (e & 7) << 2;
                int gm = tile_m + m; gm = gm > Mc ? Mc : gm;
                const float* src = A + (long)gm * lda + (k0 + kq);
                rva4[p] = *(const float4*)src;
                if (p == 0 && !tailk) __builtin_prefetch(src + 2 * BKK, 0, 3);
            }
        } else {
            #pragma unroll
            for (int p = 0; p < 16; ++p) {
                int e = (p << 8) + tid;
                int m = e >> 5, k = e & 31;
                int gm = tile_m + m; gm = gm > Mc ? Mc : gm;
                int gk = k0 + k;
                if (tailk && gk > K - 1) gk = K - 1;
                rva[p] = A[(long)gm * lda + gk];
            }
        }
        if (FLAGS & GF_BROW) {
            #pragma unroll
            for (int p = 0; p < 8; ++p) {
                int e = (p << 8) + tid;
                int n = e >> 5, k = e & 31;
                int gn = tile_n + n; gn = gn > Nc ? Nc : gn;
                int gk = k0 + k;
                if (tailk && gk > K - 1) gk = K - 1;
                rvb[p] = Bm[(long)gk * ldb + gn];
            }
        } else if (VB) {
            #pragma unroll
            for (int p = 0; p < 2; ++p) {
                int e = (p << 8) + tid;
                int n = e >> 3, kq = (e & 7) << 2;
                int gn = tile_n + n; gn = gn > Nc ? Nc : gn;
                const float* src = Bm + (long)gn * ldb + (k0 + kq);
                rvb4[p] = *(const float4*)src;
                if (p == 0 && !tailk) __builtin_prefetch(src + 2 * BKK, 0, 3);
            }
        } else {
            #pragma unroll
            for (int p = 0; p < 8; ++p) {
                int e = (p << 8) + tid;
                int n = e >> 5, k = e & 31;
                int gn = tile_n + n; gn = gn > Nc ? Nc : gn;
                int gk = k0 + k;
                if (tailk && gk > K - 1) gk = K - 1;
                rvb[p] = Bm[(long)gn * ldb + gk];
            }
        }
    };

    // Phase 2: convert + store staged registers into LDS buffer `buf`.
    auto stage_store = [&](int buf, int k0, bool tailk) {
        if ((FLAGS & GF_AT) || !VA) {
            #pragma unroll
            for (int p = 0; p < 16; ++p) {
                int e = (p << 8) + tid;
                int m = e >> 5, k = e & 31;
                float v = rva[p];
                if (tailk && (k0 + k >= K)) v = 0.f;
                As[buf][(m << 5) + k] = (_Float16)v;
            }
        } else {
            #pragma unroll
            for (int p = 0; p < 4; ++p) {
                int e = (p << 8) + tid;
                int m = e >> 3, kq = (e & 7) << 2;
                h4 h;
                if (tailk) {
                    h.x = (k0 + kq + 0 < K) ? (_Float16)rva4[p].x : (_Float16)0.f;
                    h.y = (k0 + kq + 1 < K) ? (_Float16)rva4[p].y : (_Float16)0.f;
                    h.z = (k0 + kq + 2 < K) ? (_Float16)rva4[p].z : (_Float16)0.f;
                    h.w = (k0 + kq + 3 < K) ? (_Float16)rva4[p].w : (_Float16)0.f;
                } else {
                    h.x = (_Float16)rva4[p].x; h.y = (_Float16)rva4[p].y;
                    h.z = (_Float16)rva4[p].z; h.w = (_Float16)rva4[p].w;
                }
                *(h4*)&As[buf][(m << 5) + kq] = h;
            }
        }
        if ((FLAGS & GF_BROW) || !VB) {
            #pragma unroll
            for (int p = 0; p < 8; ++p) {
                int e = (p << 8) + tid;
                int n = e >> 5, k = e & 31;
                float v = rvb[p];
                if (tailk && (k0 + k >= K)) v = 0.f;
                Bs[buf][(n << 5) + k] = (_Float16)v;
            }
        } else {
            #pragma unroll
            for (int p = 0; p < 2; ++p) {
                int e = (p << 8) + tid;
                int n = e >> 3, kq = (e & 7) << 2;
                h4 h;
                if (tailk) {
                    h.x = (k0 + kq + 0 < K) ? (_Float16)rvb4[p].x : (_Float16)0.f;
                    h.y = (k0 + kq + 1 < K) ? (_Float16)rvb4[p].y : (_Float16)0.f;
                    h.z = (k0 + kq + 2 < K) ? (_Float16)rvb4[p].z : (_Float16)0.f;
                    h.w = (k0 + kq + 3 < K) ? (_Float16)rvb4[p].w : (_Float16)0.f;
                } else {
                    h.x = (_Float16)rvb4[p].x; h.y = (_Float16)rvb4[p].y;
                    h.z = (_Float16)rvb4[p].z; h.w = (_Float16)rvb4[p].w;
                }
                *(h4*)&Bs[buf][(n << 5) + kq] = h;
            }
        }
    };

    auto compute = [&](int buf) {
        // Fragment build per ISA 7.12.2. kbA in {0,8} halves, kbB in {0,8}
        // halves -> all chunks 16B aligned: h8 (ds_load_b128) LDS loads.
        const int hl  = lane & 15;
        const int kbA = (lane >> 4) << 3;
        const int kbB = (lane >> 4) << 3;
        const _Float16* ap0 = &As[buf][((wm << 5) + hl) << 5];
        const _Float16* ap1 = ap0 + (16 << 5);
        h8 a0lo = *(const h8*)(ap0 + kbA);
        h8 a0hi = *(const h8*)(ap0 + 16 + kbA);
        h8 a1lo = *(const h8*)(ap1 + kbA);
        h8 a1hi = *(const h8*)(ap1 + 16 + kbA);
        const _Float16* bp0 = &Bs[buf][((wn << 5) + hl) << 5];
        const _Float16* bp1 = bp0 + (16 << 5);
        h8 b0lo = *(const h8*)(bp0 + kbB);
        h8 b0hi = *(const h8*)(bp0 + 16 + kbB);
        h8 b1lo = *(const h8*)(bp1 + kbB);
        h8 b1hi = *(const h8*)(bp1 + 16 + kbB);
        v16h a0 = __builtin_shufflevector(a0lo, a0hi, 0,1,2,3,4,5,6,7,8,9,10,11,12,13,14,15);
        v16h a1 = __builtin_shufflevector(a1lo, a1hi, 0,1,2,3,4,5,6,7,8,9,10,11,12,13,14,15);
        v16h b0 = __builtin_shufflevector(b0lo, b0hi, 0,1,2,3,4,5,6,7,8,9,10,11,12,13,14,15);
        v16h b1 = __builtin_shufflevector(b1lo, b1hi, 0,1,2,3,4,5,6,7,8,9,10,11,12,13,14,15);
        acc00 = __builtin_amdgcn_wmma_f32_16x16x32_f16(false, a0, false, b0, (short)0, acc00, false, false);
        acc01 = __builtin_amdgcn_wmma_f32_16x16x32_f16(false, a0, false, b1, (short)0, acc01, false, false);
        acc10 = __builtin_amdgcn_wmma_f32_16x16x32_f16(false, a1, false, b0, (short)0, acc10, false, false);
        acc11 = __builtin_amdgcn_wmma_f32_16x16x32_f16(false, a1, false, b1, (short)0, acc11, false, false);
    };

    const int ksteps = (K + 31) >> 5;
    const int kfull  = K >> 5;
    // prologue: stage k-step 0 (statically-selected tail variant)
    if (kfull > 0) { stage_load(0, false); stage_store(0, 0, false); }
    else           { stage_load(0, true);  stage_store(0, 0, true);  }
    for (int kt = 0; kt < ksteps; ++kt) {
        __syncthreads();
        const int nx = kt + 1;
        if (nx < ksteps) {
            if (nx < kfull) stage_load(nx << 5, false);
            else            stage_load(nx << 5, true);
        }
        compute(kt & 1);
        if (nx < ksteps) {
            if (nx < kfull) stage_store(nx & 1, nx << 5, false);
            else            stage_store(nx & 1, nx << 5, true);
        }
    }

    // Epilogue: C/D layout — lane = N, VGPR r = M (mb + r), mb = (lane>=16)*8
    const int hl = lane & 15;
    const int mb = (lane >> 4) << 3;
    auto out_frag = [&](const v8f& acc, int row0, int col, bool guard) {
        if (guard && col >= N) return;
        const float bv = bias ? bias[col] : 0.f;
        #pragma unroll
        for (int r = 0; r < 8; ++r) {
            int row = row0 + r;
            if (!guard || row < M) {
                float v = acc[r] * alpha + bv;
                long off = (long)row * ldc + col;
                if (FLAGS & GF_ACC)  v += C[off];
                if (FLAGS & GF_RELU) v = fmaxf(v, 0.f);
                if (FLAGS & GF_SIG)  v = 1.f / (1.f + __expf(-v));
                C[off] = v;
            }
        }
    };
    const int r0 = tile_m + (wm << 5) + mb;
    const int c0 = tile_n + (wn << 5) + hl;
    if (tile_m + BM <= M && tile_n + BN <= N) {
        out_frag(acc00, r0,      c0,      false);
        out_frag(acc01, r0,      c0 + 16, false);
        out_frag(acc10, r0 + 16, c0,      false);
        out_frag(acc11, r0 + 16, c0 + 16, false);
    } else {
        out_frag(acc00, r0,      c0,      true);
        out_frag(acc01, r0,      c0 + 16, true);
        out_frag(acc10, r0 + 16, c0,      true);
        out_frag(acc11, r0 + 16, c0 + 16, true);
    }
}

// ---------------------------------------------------------------------------
// Row softmax (one block per row)
// ---------------------------------------------------------------------------
__global__ __launch_bounds__(256)
void hamlet_softmax_rows(float* __restrict__ S, int len)
{
    __shared__ float red[256];
    float* p = S + (long)blockIdx.x * len;
    int tid = threadIdx.x;
    float m = -1e30f;
    for (int i = tid; i < len; i += 256) m = fmaxf(m, p[i]);
    red[tid] = m; __syncthreads();
    for (int s = 128; s > 0; s >>= 1) {
        if (tid < s) red[tid] = fmaxf(red[tid], red[tid + s]);
        __syncthreads();
    }
    m = red[0]; __syncthreads();
    float sum = 0.f;
    for (int i = tid; i < len; i += 256) { float e = __expf(p[i] - m); p[i] = e; sum += e; }
    red[tid] = sum; __syncthreads();
    for (int s = 128; s > 0; s >>= 1) {
        if (tid < s) red[tid] += red[tid + s];
        __syncthreads();
    }
    float inv = 1.f / red[0];
    for (int i = tid; i < len; i += 256) p[i] *= inv;
}

// e += gate * ctx
__global__ void hamlet_gate_update(float* __restrict__ e, const float* __restrict__ g,
                                   const float* __restrict__ c, long n)
{
    long i = (long)blockIdx.x * blockDim.x + threadIdx.x;
    if (i < n) e[i] += g[i] * c[i];
}

// spatial mean of token-major [B*S,256] -> [B,256]
__global__ __launch_bounds__(256)
void hamlet_mean(const float* __restrict__ e, float* __restrict__ out, int S)
{
    int b = blockIdx.x, c = threadIdx.x;
    const float* p = e + ((long)b * S << 8) + c;
    float s = 0.f;
    for (int i = 0; i < S; ++i) s += p[(long)i << 8];
    out[(b << 8) + c] = s / (float)S;
}

// tiny Linear (M<=8): Y[m,n] = act(sum_k X[m,k]*W[n,k] + b[n])
__global__ void hamlet_small_linear(const float* __restrict__ X, const float* __restrict__ W,
                                    const float* __restrict__ bias, float* __restrict__ Y,
                                    int Mm, int Nn, int Kk, int relu)
{
    int i = blockIdx.x * blockDim.x + threadIdx.x;
    if (i >= Mm * Nn) return;
    int m = i / Nn, n = i % Nn;
    float s = bias ? bias[n] : 0.f;
    for (int k = 0; k < Kk; ++k) s += X[m * Kk + k] * W[n * Kk + k];
    if (relu) s = fmaxf(s, 0.f);
    Y[i] = s;
}

// depthwise 3x3, per-(sample,group) kernel; token-major [B*S,256]
__global__ __launch_bounds__(256)
void hamlet_dwconv(const float* __restrict__ xt, const float* __restrict__ kern,
                   float* __restrict__ y, int H, int W)
{
    const int S = H * W;
    long i = (long)blockIdx.x * 256 + threadIdx.x;
    long total = (long)8 * S * 256;
    if (i >= total) return;
    int c = (int)(i & 255);
    long t = i >> 8;
    int b = (int)(t / S);
    int s = (int)(t % S);
    int h = s / W, w = s % W;
    const float* kk = kern + ((b << 3) + (c >> 5)) * 9;  // group = c / 32
    float acc = 0.f;
    #pragma unroll
    for (int dy = -1; dy <= 1; ++dy) {
        #pragma unroll
        for (int dx = -1; dx <= 1; ++dx) {
            int hh = h + dy, ww = w + dx;
            if (hh >= 0 && hh < H && ww >= 0 && ww < W)
                acc += kk[(dy + 1) * 3 + (dx + 1)] * xt[(((long)b * S + hh * W + ww) << 8) + c];
        }
    }
    y[i] = acc;
}

// per-channel sum & sumsq over T tokens (token-major [T,256])
__global__ __launch_bounds__(256)
void hamlet_bn_stats(const float* __restrict__ y, float* __restrict__ sums, int T)
{
    __shared__ float s1[256], s2[256];
    int c = blockIdx.x;
    float a = 0.f, b = 0.f;
    for (int t = threadIdx.x; t < T; t += 256) {
        float v = y[((long)t << 8) + c];
        a += v; b += v * v;
    }
    s1[threadIdx.x] = a; s2[threadIdx.x] = b; __syncthreads();
    for (int s = 128; s > 0; s >>= 1) {
        if (threadIdx.x < s) { s1[threadIdx.x] += s1[threadIdx.x + s]; s2[threadIdx.x] += s2[threadIdx.x + s]; }
        __syncthreads();
    }
    if (threadIdx.x == 0) { sums[c] = s1[0]; sums[256 + c] = s2[0]; }
}

// out[b,c,soff+s] = xt + BN(y2)*gamma + beta ; token-major -> [B,256,TOT]
__global__ __launch_bounds__(256)
void hamlet_bn_apply_out(const float* __restrict__ xt, const float* __restrict__ y2,
                         const float* __restrict__ sums, const float* __restrict__ gamma,
                         const float* __restrict__ beta, float* __restrict__ out,
                         int T, int S, int soff, int TOT)
{
    long i = (long)blockIdx.x * 256 + threadIdx.x;
    if (i >= (long)T * 256) return;
    int c = (int)(i & 255);
    long t = i >> 8;
    int b = (int)(t / S);
    int s = (int)(t % S);
    float mu  = sums[c] / (float)T;
    float var = sums[256 + c] / (float)T - mu * mu;
    float v = (y2[i] - mu) * rsqrtf(var + 1e-5f) * gamma[c] + beta[c];
    out[((long)b * 256 + c) * TOT + soff + s] = xt[i] + v;
}

// ---------------------------------------------------------------------------
// Dispatch only the reachable (FLAGS, VA, VB) combos (10 instantiations):
//   GF_AT        -> va always false (A is [K,M]);         vb runtime
//   GF_BROW      -> vb always false (B is row-major);     va runtime
//   0 / ACC|RELU / SIG -> our call sites always have va=vb=true; keep a
//                         scalar-scalar fallback for safety.
// ---------------------------------------------------------------------------
#define HAMLET_LAUNCH(F, VAV, VBV) \
    hamlet_gemm_wmma<F, VAV, VBV><<<grid, 256, 0, stream>>>( \
        A, Bm, bias, C, M, N, K, lda, ldb, ldc, sAz, sBz, sCz, alpha)

static void launch_gemm(const float* A, const float* Bm, const float* bias, float* C,
                        int M, int N, int K, int lda, int ldb, int ldc,
                        long sAz, long sBz, long sCz, int Z, float alpha, int flags,
                        hipStream_t stream)
{
    dim3 grid((N + BN - 1) / BN, (M + BM - 1) / BM, Z);
    const bool va = !(flags & GF_AT)   && ((lda & 3) == 0) && ((sAz & 3) == 0) &&
                    ((((unsigned long)A) & 15) == 0);
    const bool vb = !(flags & GF_BROW) && ((ldb & 3) == 0) && ((sBz & 3) == 0) &&
                    ((((unsigned long)Bm) & 15) == 0);
    switch (flags) {
    case GF_AT:
        if (vb) HAMLET_LAUNCH(GF_AT, false, true);
        else    HAMLET_LAUNCH(GF_AT, false, false);
        break;
    case GF_BROW:
        if (va) HAMLET_LAUNCH(GF_BROW, true, false);
        else    HAMLET_LAUNCH(GF_BROW, false, false);
        break;
    case GF_ACC | GF_RELU:
        if (va && vb) HAMLET_LAUNCH(GF_ACC | GF_RELU, true, true);
        else          HAMLET_LAUNCH(GF_ACC | GF_RELU, false, false);
        break;
    case GF_SIG:
        if (va && vb) HAMLET_LAUNCH(GF_SIG, true, true);
        else          HAMLET_LAUNCH(GF_SIG, false, false);
        break;
    default:
        if (va && vb) HAMLET_LAUNCH(0, true, true);
        else          HAMLET_LAUNCH(0, false, false);
        break;
    }
}

extern "C" void kernel_launch(void* const* d_in, const int* in_sizes, int n_in,
                              void* d_out, int out_size, void* d_ws, size_t ws_size,
                              hipStream_t stream)
{
    (void)in_sizes; (void)n_in; (void)out_size; (void)ws_size;

    const float* f[4]  = {(const float*)d_in[0], (const float*)d_in[3], (const float*)d_in[6], (const float*)d_in[9]};
    const float* Wp[4] = {(const float*)d_in[1], (const float*)d_in[4], (const float*)d_in[7], (const float*)d_in[10]};
    const float* bp[4] = {(const float*)d_in[2], (const float*)d_in[5], (const float*)d_in[8], (const float*)d_in[11]};
    const float* Wq  = (const float*)d_in[12];
    const float* Wk  = (const float*)d_in[13];
    const float* Wv  = (const float*)d_in[14];
    const float* Wo  = (const float*)d_in[15];
    const float* Wg1 = (const float*)d_in[16];   // [256,512]
    const float* Wg2 = (const float*)d_in[17];
    const float* Wcp = (const float*)d_in[18];
    const float* Wkg1 = (const float*)d_in[19];  // [512,256]
    const float* Wkg2 = (const float*)d_in[20];  // [72,512]
    const float* Wit = (const float*)d_in[21];
    const float* Wot = (const float*)d_in[22];
    const float* bq  = (const float*)d_in[23];
    const float* bk  = (const float*)d_in[24];
    const float* bv  = (const float*)d_in[25];
    const float* bo  = (const float*)d_in[26];
    const float* bg1 = (const float*)d_in[27];
    const float* bg2 = (const float*)d_in[28];
    const float* bcp = (const float*)d_in[29];
    const float* bkg1 = (const float*)d_in[30];
    const float* bkg2 = (const float*)d_in[31];
    const float* bit = (const float*)d_in[32];
    const float* bot = (const float*)d_in[33];
    const float* gamma = (const float*)d_in[34];
    const float* beta  = (const float*)d_in[35];
    float* out = (float*)d_out;

    const int DIM[4] = {64, 128, 256, 512};
    const int SZ[4]  = {56, 28, 14, 7};
    int S[4], offT[4], offS[4];
    int acc_t = 0, acc_s = 0;
    for (int i = 0; i < 4; ++i) {
        S[i] = SZ[i] * SZ[i];
        offT[i] = acc_t; acc_t += 8 * S[i];
        offS[i] = acc_s; acc_s += S[i];
    }
    const int T_all = acc_t;          // 33320
    const int TOT   = acc_s;          // 4165
    const size_t MAXT = (size_t)8 * S[0];   // 25088 tokens

    float* ws = (float*)d_ws;
    size_t o = 0;
    auto alloc = [&](size_t n) { float* p = ws + o; o += n; return p; };
    float* flat   = alloc((size_t)T_all * 256);
    float* ebuf   = alloc((size_t)T_all * 256);
    float* qb     = alloc(MAXT * 256);
    float* kbuf   = alloc(MAXT * 256);
    float* vbuf   = alloc(MAXT * 256);
    float* ctxb   = alloc(MAXT * 256);
    float* ctxo   = alloc(MAXT * 256);
    float* g1b    = alloc(MAXT * 256);
    float* gateb  = alloc(MAXT * 256);
    float* scores = alloc((size_t)8 * S[0] * S[1]);   // per-batch [H, Si, Sj] max
    float* xtb    = alloc(MAXT * 256);
    float* yb     = alloc(MAXT * 256);
    float* y2b    = alloc(MAXT * 256);
    float* meanb  = alloc(8 * 256);
    float* cvec   = alloc(8 * 256);
    float* kg1b   = alloc(8 * 512);
    float* kernb  = alloc(8 * 72);
    float* bnsums = alloc(512);

    auto gemm = [&](const float* A, const float* Bm, const float* bias, float* C,
                    int M, int N, int K, int lda, int ldb, int ldc,
                    long sAz, long sBz, long sCz, int Z, float alpha, int flags) {
        launch_gemm(A, Bm, bias, C, M, N, K, lda, ldb, ldc, sAz, sBz, sCz, Z, alpha, flags, stream);
    };

    // ---- 1x1 projections: flat[b, s, :] = Wp_i @ f_i[b, :, s] + bp_i ----
    for (int i = 0; i < 4; ++i) {
        gemm(f[i], Wp[i], bp[i], flat + (size_t)offT[i] * 256,
             S[i], 256, DIM[i], S[i], DIM[i], 256,
             (long)DIM[i] * S[i], 0, (long)S[i] * 256, 8, 1.f, GF_AT);
    }
    hipMemcpyAsync(ebuf, flat, (size_t)T_all * 256 * sizeof(float),
                   hipMemcpyDeviceToDevice, stream);

    const float attn_scale = 0.17677669529663687f;   // 1/sqrt(32)

    // ---- ACSA cross-scale attention ----
    for (int i = 0; i < 4; ++i) {
        const int Mi = 8 * S[i];
        gemm(flat + (size_t)offT[i] * 256, Wq, bq, qb, Mi, 256, 256, 256, 256, 256,
             0, 0, 0, 1, 1.f, 0);
        for (int j = 0; j < 4; ++j) {
            if (j == i) continue;
            const int Mj = 8 * S[j];
            gemm(flat + (size_t)offT[j] * 256, Wk, bk, kbuf, Mj, 256, 256, 256, 256, 256,
                 0, 0, 0, 1, 1.f, 0);
            gemm(flat + (size_t)offT[j] * 256, Wv, bv, vbuf, Mj, 256, 256, 256, 256, 256,
                 0, 0, 0, 1, 1.f, 0);
            for (int b = 0; b < 8; ++b) {
                // scores[h, qi, kj] = scale * Q_bh @ K_bh^T   (Z = heads)
                gemm(qb + (size_t)b * S[i] * 256, kbuf + (size_t)b * S[j] * 256, nullptr,
                     scores, S[i], S[j], 32, 256, 256, S[j],
                     32, 32, (long)S[i] * S[j], 8, attn_scale, 0);
                hamlet_softmax_rows<<<dim3(8 * S[i]), 256, 0, stream>>>(scores, S[j]);
                // ctx[h, qi, :] = attn @ V_bh, scattered into [tokens, 256] at h*32
                gemm(scores, vbuf + (size_t)b * S[j] * 256, nullptr,
                     ctxb + (size_t)b * S[i] * 256, S[i], 32, S[j], S[j], 256, 256,
                     (long)S[i] * S[j], 32, 32, 8, 1.f, GF_BROW);
            }
            // ctx -> Wo
            gemm(ctxb, Wo, bo, ctxo, Mi, 256, 256, 256, 256, 256, 0, 0, 0, 1, 1.f, 0);
            // gate MLP: relu(e@Wg1[:, :256]^T + ctx@Wg1[:, 256:]^T + bg1)
            gemm(ebuf + (size_t)offT[i] * 256, Wg1, bg1, g1b, Mi, 256, 256, 256, 512, 256,
                 0, 0, 0, 1, 1.f, 0);
            gemm(ctxo, Wg1 + 256, nullptr, g1b, Mi, 256, 256, 256, 512, 256,
                 0, 0, 0, 1, 1.f, GF_ACC | GF_RELU);
            gemm(g1b, Wg2, bg2, gateb, Mi, 256, 256, 256, 256, 256,
                 0, 0, 0, 1, 1.f, GF_SIG);
            long n = (long)Mi * 256;
            hamlet_gate_update<<<dim3((unsigned)((n + 255) / 256)), 256, 0, stream>>>(
                ebuf + (size_t)offT[i] * 256, gateb, ctxo, n);
        }
    }

    // ---- CECM per scale ----
    for (int i = 0; i < 4; ++i) {
        const int Mi = 8 * S[i];
        const float* e_i = ebuf + (size_t)offT[i] * 256;
        hamlet_mean<<<8, 256, 0, stream>>>(e_i, meanb, S[i]);
        hamlet_small_linear<<<(8 * 256 + 255) / 256, 256, 0, stream>>>(meanb, Wcp, bcp, cvec, 8, 256, 256, 1);
        hamlet_small_linear<<<(8 * 512 + 255) / 256, 256, 0, stream>>>(cvec, Wkg1, bkg1, kg1b, 8, 512, 256, 1);
        hamlet_small_linear<<<(8 * 72 + 255) / 256, 256, 0, stream>>>(kg1b, Wkg2, bkg2, kernb, 8, 72, 512, 0);
        // xt = conv1x1(x, Wit)
        gemm(e_i, Wit, bit, xtb, Mi, 256, 256, 256, 256, 256, 0, 0, 0, 1, 1.f, 0);
        long n = (long)Mi * 256;
        hamlet_dwconv<<<dim3((unsigned)((n + 255) / 256)), 256, 0, stream>>>(xtb, kernb, yb, SZ[i], SZ[i]);
        // y = conv1x1(y, Wot)
        gemm(yb, Wot, bot, y2b, Mi, 256, 256, 256, 256, 256, 0, 0, 0, 1, 1.f, 0);
        hamlet_bn_stats<<<256, 256, 0, stream>>>(y2b, bnsums, Mi);
        hamlet_bn_apply_out<<<dim3((unsigned)((n + 255) / 256)), 256, 0, stream>>>(
            xtb, y2b, bnsums, gamma, beta, out, Mi, S[i], offS[i], TOT);
    }
}
